// Gumbel_Token_Selection_Block_Wrapper_3118146257251
// MI455X (gfx1250) — matrix operand order repaired and checked by reading, not compiled, not run on hardware
//
#include <hip/hip_runtime.h>
#include <hip/hip_bf16.h>
#include <math.h>

typedef __bf16 bf16;
typedef float    v8f   __attribute__((ext_vector_type(8)));
typedef bf16     v16bf __attribute__((ext_vector_type(16)));
typedef unsigned v4u   __attribute__((ext_vector_type(4)));
typedef int      v8i   __attribute__((ext_vector_type(8)));
typedef int      v4i   __attribute__((ext_vector_type(4)));

#define B_    32
#define N_    577
#define NPAD  608            // 19 * 32, padded token count for attention
#define D_    768
#define H_    12
#define HD_   64
#define DFF_  3072
#define NTOK  (B_ * N_)      // 18464 (multiple of 16)
#define NSEL  289
#define KTOP  288

union FragAB { v16bf v; unsigned u[8]; bf16 h[16]; };

// DPP16 row_ror rotate-reductions within each 16-lane row (pure VALU, no LDS)
__device__ __forceinline__ float rowred_max16(float v) {
  int x;
  x = __builtin_amdgcn_update_dpp(__float_as_int(v), __float_as_int(v),
                                  0x121, 0xf, 0xf, false);
  v = fmaxf(v, __int_as_float(x));
  x = __builtin_amdgcn_update_dpp(__float_as_int(v), __float_as_int(v),
                                  0x122, 0xf, 0xf, false);
  v = fmaxf(v, __int_as_float(x));
  x = __builtin_amdgcn_update_dpp(__float_as_int(v), __float_as_int(v),
                                  0x124, 0xf, 0xf, false);
  v = fmaxf(v, __int_as_float(x));
  x = __builtin_amdgcn_update_dpp(__float_as_int(v), __float_as_int(v),
                                  0x128, 0xf, 0xf, false);
  return fmaxf(v, __int_as_float(x));
}
__device__ __forceinline__ float rowred_sum16(float v) {
  int x;
  x = __builtin_amdgcn_update_dpp(__float_as_int(v), __float_as_int(v),
                                  0x121, 0xf, 0xf, false);
  v += __int_as_float(x);
  x = __builtin_amdgcn_update_dpp(__float_as_int(v), __float_as_int(v),
                                  0x122, 0xf, 0xf, false);
  v += __int_as_float(x);
  x = __builtin_amdgcn_update_dpp(__float_as_int(v), __float_as_int(v),
                                  0x124, 0xf, 0xf, false);
  v += __int_as_float(x);
  x = __builtin_amdgcn_update_dpp(__float_as_int(v), __float_as_int(v),
                                  0x128, 0xf, 0xf, false);
  return v + __int_as_float(x);
}

// ---------------------------------------------------------------------------
// TDM 2-D tile load to LDS (D# per CDNA5 ISA).
// dim0 = contiguous elements (2-byte), dim1 = rows with stride_d0 elements.
// Rows beyond tensor_d1 are zero-filled by hardware. Optional LDS padding:
// pad_amount+1 DWORDs inserted every 2^(pad_interval+1) DWORDs.
// This toolchain exposes the 6-arg builtin; the trailing group operands are
// zero here (2-D tensor, single descriptor, no gather/iterate).
// ---------------------------------------------------------------------------
__device__ __forceinline__ void tdm_load_2d_b16(
    unsigned lds_addr, const void* gaddr,
    unsigned tile_d0, unsigned tile_d1,
    unsigned tensor_d0, unsigned tensor_d1,
    unsigned stride_d0, unsigned pad_interval, unsigned pad_amount,
    bool pad_en) {
  unsigned long long ga = (unsigned long long)(uintptr_t)gaddr;
  v4u g0;
  g0[0] = 1u;                                        // count=1, user mode
  g0[1] = lds_addr;                                  // LDS byte address
  g0[2] = (unsigned)(ga & 0xffffffffu);              // global_addr[31:0]
  g0[3] = (unsigned)((ga >> 32) & 0x1ffffffu) | (2u << 30); // [56:32] | type=2
  v8i g1;
  unsigned d0 = (1u << 16)                           // data_size = 2 bytes
              | (pad_en ? ((1u << 20) | (pad_interval << 22) | (pad_amount << 25))
                        : 0u);
  g1[0] = (int)d0;
  g1[1] = (int)((tensor_d0 & 0xffffu) << 16);        // tensor_dim0[15:0]
  g1[2] = (int)((tensor_d0 >> 16) | ((tensor_d1 & 0xffffu) << 16));
  g1[3] = (int)((tensor_d1 >> 16) | (tile_d0 << 16)); // tile_dim0
  g1[4] = (int)(tile_d1 & 0xffffu);                  // tile_dim1 (tile_dim2=0)
  g1[5] = (int)stride_d0;                            // tensor_dim0_stride[31:0]
  g1[6] = 0;                                         // stride hi / dim1_stride lo
  g1[7] = 0;
  v4i g2 = {0, 0, 0, 0};                             // 2-D tensor: unused
  v4i g3 = {0, 0, 0, 0};
  v8i g4 = {0, 0, 0, 0, 0, 0, 0, 0};
  __builtin_amdgcn_tensor_load_to_lds(g0, g1, g2, g3, g4, 0);
}

// ---------------------------------------------------------------------------
// f32 -> bf16 convert (weights)
// ---------------------------------------------------------------------------
__global__ __launch_bounds__(256) void cvt_k(const float* __restrict__ in,
                                             bf16* __restrict__ out, int n) {
  int i = blockIdx.x * blockDim.x + threadIdx.x;
  int stride = gridDim.x * blockDim.x;
  for (; i < n; i += stride) out[i] = (bf16)in[i];
}

// ---------------------------------------------------------------------------
// LayerNorm: f32 [row][768] -> bf16, one block per row
// ---------------------------------------------------------------------------
__global__ __launch_bounds__(256) void ln_k(const float* __restrict__ x,
                                            const float* __restrict__ w,
                                            const float* __restrict__ b,
                                            bf16* __restrict__ out) {
  int row = blockIdx.x;
  const float* xr = x + (size_t)row * D_;
  __shared__ float s1[256], s2[256];
  float v[3];
  float lsum = 0.f, lsq = 0.f;
#pragma unroll
  for (int t = 0; t < 3; t++) {
    v[t] = xr[threadIdx.x + t * 256];
    lsum += v[t];
    lsq  += v[t] * v[t];
  }
  s1[threadIdx.x] = lsum; s2[threadIdx.x] = lsq;
  __syncthreads();
  for (int off = 128; off > 0; off >>= 1) {
    if (threadIdx.x < off) {
      s1[threadIdx.x] += s1[threadIdx.x + off];
      s2[threadIdx.x] += s2[threadIdx.x + off];
    }
    __syncthreads();
  }
  float mean = s1[0] * (1.f / D_);
  float var  = s2[0] * (1.f / D_) - mean * mean;
  float rstd = rsqrtf(var + 1e-6f);
#pragma unroll
  for (int t = 0; t < 3; t++) {
    int c = threadIdx.x + t * 256;
    out[(size_t)row * D_ + c] = (bf16)((v[t] - mean) * rstd * w[c] + b[c]);
  }
}

// ---------------------------------------------------------------------------
// bf16 WMMA GEMM: C[M][NO] = A[M][K] @ Bw[K][NO] (+bias) with fused epilogue
//   EPI 0: outH = bf16(acc + bias)
//   EPI 1: outF = xres + ls * (acc + bias)         (residual+layerscale)
//   EPI 2: outH = bf16(gelu_exact(acc + bias))
// Block: 256 thr (8 waves) -> 128x64 tile; wave (wm,wn) in 4x2 -> 32x32.
// A tile staged by the Tensor Data Mover (wave 0 issues; HW zero-fills
// rows >= M and inserts the 16B/row bank padding). B staged manually
// (transposed) since TDM cannot transpose.
// ---------------------------------------------------------------------------
template <int EPI>
__global__ __launch_bounds__(256)
void gemm_bf16_k(const bf16* __restrict__ A, const bf16* __restrict__ Bw,
                 const float* __restrict__ bias,
                 const float* __restrict__ xres, const float* __restrict__ ls,
                 bf16* __restrict__ outH, float* __restrict__ outF,
                 int M, int K, int NO) {
  __shared__ bf16 As[128][40];   // rows x (32 K + 8 pad) — written by TDM
  __shared__ bf16 Bs[64][40];    // transposed: [n][k]
  const int tid  = threadIdx.x;
  const int lane = tid & 31, wave = tid >> 5;
  const int wm = wave >> 1, wn = wave & 1;
  const int rowBase = blockIdx.y * 128;
  const int colBase = blockIdx.x * 64;
  const int l15 = lane & 15;

  v8f acc[2][2];
#pragma unroll
  for (int i = 0; i < 2; i++)
#pragma unroll
    for (int j = 0; j < 2; j++)
#pragma unroll
      for (int e = 0; e < 8; e++) acc[i][j][e] = 0.f;

  const int bk = tid >> 3;           // 0..31
  const int bn = (tid & 7) * 8;      // 0..56
  const unsigned asLds = (unsigned)(size_t)(void*)&As[0][0];
  const unsigned tenD1 = (unsigned)(M - rowBase);   // HW zero-fills rows >= M

  for (int k0 = 0; k0 < K; k0 += 32) {
    // --- A tile via TDM: 128 rows x 32 bf16, +16B pad per 64B row ---
    if (wave == 0) {
      tdm_load_2d_b16(asLds, A + (size_t)rowBase * K + k0,
                      /*tile_d0=*/32, /*tile_d1=*/128,
                      /*tensor_d0=*/(unsigned)(K - k0), /*tensor_d1=*/tenD1,
                      /*stride_d0=*/(unsigned)K,
                      /*pad_interval=*/3, /*pad_amount=*/3, /*pad_en=*/true);
      __builtin_amdgcn_s_wait_tensorcnt(0);
    }
    // --- B tile: 32x64 bf16, transposed into Bs[n][k] ---
    {
      int4 val = *reinterpret_cast<const int4*>(
          Bw + (size_t)(k0 + bk) * NO + colBase + bn);
      union { int4 i4; bf16 h[8]; } u; u.i4 = val;
#pragma unroll
      for (int e = 0; e < 8; e++) Bs[bn + e][bk] = u.h[e];
    }
    // prefetch next k-step of B (near-cache)
    {
      int kn = min(k0 + 32, K - 32);
      __builtin_prefetch(Bw + (size_t)(kn + bk) * NO + colBase + bn, 0, 3);
    }
    __syncthreads();

    // --- fragments (contiguous k = 16*half + s in both A and B) ---
    FragAB af[2], bfr[2];
#pragma unroll
    for (int i = 0; i < 2; i++) {
      int mrow = wm * 32 + i * 16 + l15;
      int kb = (lane < 16) ? 0 : 16;
#pragma unroll
      for (int j = 0; j < 8; j++) {
        int koff = kb + 2 * j;
        af[i].u[j] = *reinterpret_cast<const unsigned*>(&As[mrow][koff]);
      }
    }
#pragma unroll
    for (int j = 0; j < 2; j++) {
      int ncol = wn * 32 + j * 16 + l15;
      int kb = (lane < 16) ? 0 : 16;
#pragma unroll
      for (int q = 0; q < 8; q++)
        bfr[j].u[q] = *reinterpret_cast<const unsigned*>(&Bs[ncol][kb + 2 * q]);
    }
#pragma unroll
    for (int i = 0; i < 2; i++)
#pragma unroll
      for (int j = 0; j < 2; j++)
        acc[i][j] = __builtin_amdgcn_wmma_f32_16x16x32_bf16(
            false, af[i].v, false, bfr[j].v, (short)0, acc[i][j], false, false);
    __syncthreads();
  }

  // --- epilogue ---
  const int rsub = (lane < 16) ? 0 : 8;
#pragma unroll
  for (int i = 0; i < 2; i++) {
#pragma unroll
    for (int j = 0; j < 2; j++) {
      int col = colBase + wn * 32 + j * 16 + l15;
      float bcol = bias ? bias[col] : 0.f;
#pragma unroll
      for (int e = 0; e < 8; e++) {
        int row = rowBase + wm * 32 + i * 16 + e + rsub;
        if (row >= M) continue;
        float v = acc[i][j][e] + bcol;
        size_t idx = (size_t)row * NO + col;
        if (EPI == 0) {
          outH[idx] = (bf16)v;
        } else if (EPI == 1) {
          outF[idx] = xres[idx] + ls[col] * v;
        } else {
          float g = 0.5f * v * (1.f + erff(v * 0.70710678118f));
          outH[idx] = (bf16)g;
        }
      }
    }
  }
}

// ---------------------------------------------------------------------------
// V transpose: qkv bf16 [B][N][2304] (v at +1536) -> vT [B][H][64][NPAD],
// zero-padded along the token dim. One block per (b,h); coalesced writes.
// ---------------------------------------------------------------------------
__global__ __launch_bounds__(256)
void vtr_k(const bf16* __restrict__ qkv, bf16* __restrict__ vT) {
  int b = blockIdx.x / H_;
  int h = blockIdx.x % H_;
  const size_t qkvB = (size_t)b * N_ * 2304;
  bf16* dst = vT + ((size_t)(b * H_ + h)) * HD_ * NPAD;
  for (int hd = 0; hd < HD_; hd++) {
    for (int tok = threadIdx.x; tok < NPAD; tok += 256) {
      bf16 v = (tok < N_)
          ? qkv[qkvB + (size_t)tok * 2304 + 1536 + h * HD_ + hd] : (bf16)0.f;
      dst[(size_t)hd * NPAD + tok] = v;
    }
  }
}

// ---------------------------------------------------------------------------
// Flash attention. qkv bf16 [B][N][2304] (q|k per head), vT [B][H][64][NPAD],
// ctx bf16 [B][N][768]. One wave = 16 query rows of one (b,h). All inner-loop
// loads unconditional; padding masked only in the S logits (v_cndmask).
// ---------------------------------------------------------------------------
__global__ __launch_bounds__(256)
void attn_k(const bf16* __restrict__ qkv, const bf16* __restrict__ vT,
            bf16* __restrict__ ctx) {
  __shared__ bf16 Pb[8][16][34];   // per-wave P tile (16 rows x 32 cols + pad)
  int bid = blockIdx.x;
  int b = bid / (H_ * 5);
  int rem = bid % (H_ * 5);
  int h = rem / 5;
  int mblk = rem % 5;
  int lane = threadIdx.x & 31, wave = threadIdx.x >> 5;
  int mbase = mblk * 128 + wave * 16;
  if (mbase >= N_) return;           // wave-uniform exit; no block syncs used

  const size_t qkvB = (size_t)b * N_ * 2304;
  const int l15 = lane & 15;
  const int hi = (lane < 16) ? 0 : 1;
  const int kbB = hi ? 16 : 0;       // B-fragment K-half select
  const int kbA = hi ? 16 : 0;       // A-fragment K-base select

  // Q A-fragments for K-dim steps 0 (k 0-31) and 1 (k 32-63)
  FragAB aq[2];
  {
    const bf16* qr = qkv + qkvB + (size_t)(mbase + l15) * 2304 + h * HD_;
#pragma unroll
    for (int s = 0; s < 2; s++)
#pragma unroll
      for (int j = 0; j < 8; j++) {
        int kk = s * 32 + kbA + 2 * j;
        aq[s].u[j] = *reinterpret_cast<const unsigned*>(qr + kk);
      }
  }

  float m_run[8], l_run[8];          // l_run is a per-lane PARTIAL sum
  v8f o[4];
#pragma unroll
  for (int e = 0; e < 8; e++) { m_run[e] = -3.0e38f; l_run[e] = 0.f; }
#pragma unroll
  for (int t = 0; t < 4; t++)
#pragma unroll
    for (int e = 0; e < 8; e++) o[t][e] = 0.f;

  const bf16* vbase = vT + ((size_t)(b * H_ + h)) * HD_ * NPAD;
  const bf16* krow  = qkv + qkvB + 768 + h * HD_;

  for (int jb = 0; jb < 19; jb++) { // 19 * 32 = 608 columns
    // ---- issue all K fragment loads (4 fragments, contiguous 32B each) ----
    FragAB bk[2][2];                // [jj][kstep]
#pragma unroll
    for (int jj = 0; jj < 2; jj++) {
      const bf16* kr = krow + (size_t)(jb * 32 + jj * 16 + l15) * 2304;
#pragma unroll
      for (int q = 0; q < 8; q++) {
        bk[jj][0].u[q] = *reinterpret_cast<const unsigned*>(kr + kbB + 2 * q);
        bk[jj][1].u[q] = *reinterpret_cast<const unsigned*>(kr + 32 + kbB + 2 * q);
      }
    }
    // ---- S = Q @ K^T ----
    v8f sacc[2];
#pragma unroll
    for (int jj = 0; jj < 2; jj++) {
      v8f z;
#pragma unroll
      for (int e = 0; e < 8; e++) z[e] = 0.f;
      z = __builtin_amdgcn_wmma_f32_16x16x32_bf16(
          false, aq[0].v, false, bk[jj][0].v, (short)0, z, false, false);
      sacc[jj] = __builtin_amdgcn_wmma_f32_16x16x32_bf16(
          false, aq[1].v, false, bk[jj][1].v, (short)0, z, false, false);
    }
    // ---- issue V fragment loads early (consumed after softmax) ----
    FragAB vf[4];
#pragma unroll
    for (int t = 0; t < 4; t++) {
      const bf16* vr = vbase + (size_t)(t * 16 + l15) * NPAD + jb * 32 + kbB;
#pragma unroll
      for (int q = 0; q < 8; q++)
        vf[t].u[q] = *reinterpret_cast<const unsigned*>(vr + 2 * q);
    }
    // prefetch next block's K rows (near-cache)
    if (jb < 18) {
      __builtin_prefetch(krow + (size_t)(jb * 32 + 32 + l15) * 2304, 0, 3);
      __builtin_prefetch(krow + (size_t)(jb * 32 + 48 + l15) * 2304, 0, 3);
    }

    // ---- scale + column mask (branch-free) ----
#pragma unroll
    for (int jj = 0; jj < 2; jj++) {
      bool vn = (jb * 32 + jj * 16 + l15) < N_;
#pragma unroll
      for (int e = 0; e < 8; e++)
        sacc[jj][e] = vn ? sacc[jj][e] * 0.125f : -1e30f;
    }

    // ---- online softmax; row of element e is (e + 8*hi) ----
    float corr[8];
#pragma unroll
    for (int e = 0; e < 8; e++) {
      float loc = rowred_max16(fmaxf(sacc[0][e], sacc[1][e]));
      float mnew = fmaxf(m_run[e], loc);
      corr[e] = expf(m_run[e] - mnew);
      float p0 = expf(sacc[0][e] - mnew);
      float p1 = expf(sacc[1][e] - mnew);
      int prow = e + 8 * hi;
      Pb[wave][prow][l15]      = (bf16)p0;
      Pb[wave][prow][16 + l15] = (bf16)p1;
      l_run[e] = l_run[e] * corr[e] + (p0 + p1);
      m_run[e] = mnew;
    }

    // ---- P C-layout -> A-layout via per-wave LDS ----
    FragAB pf;
#pragma unroll
    for (int j = 0; j < 8; j++) {
      int kk = kbA + 2 * j;
      pf.u[j] = *reinterpret_cast<const unsigned*>(&Pb[wave][l15][kk]);
    }

    // ---- O = O*corr + P @ V ----
#pragma unroll
    for (int t = 0; t < 4; t++) {
#pragma unroll
      for (int e = 0; e < 8; e++) o[t][e] *= corr[e];
      o[t] = __builtin_amdgcn_wmma_f32_16x16x32_bf16(
          false, pf.v, false, vf[t].v, (short)0, o[t], false, false);
    }
  }

  // one-time cross-lane reduction of the partial row sums
  float linv[8];
#pragma unroll
  for (int e = 0; e < 8; e++) linv[e] = 1.f / rowred_sum16(l_run[e]);

#pragma unroll
  for (int t = 0; t < 4; t++) {
    int hd = t * 16 + l15;
#pragma unroll
    for (int e = 0; e < 8; e++) {
      int row = mbase + e + 8 * hi;
      if (row < N_) {
        float v = o[t][e] * linv[e];
        ctx[((size_t)b * N_ + row) * D_ + h * HD_ + hd] = (bf16)v;
      }
    }
  }
}

// ---------------------------------------------------------------------------
// CLS attention row (row 0) per (b,h): softmax + mean over heads via atomics
// ---------------------------------------------------------------------------
__global__ __launch_bounds__(256)
void clsattn_k(const bf16* __restrict__ qkv, float* __restrict__ cls) {
  int b = blockIdx.x / H_;
  int h = blockIdx.x % H_;
  __shared__ float q0[64];
  __shared__ float sl[N_];
  __shared__ float red[256];
  const size_t qkvB = (size_t)b * N_ * 2304;
  if (threadIdx.x < 64)
    q0[threadIdx.x] = (float)qkv[qkvB + h * HD_ + threadIdx.x];
  __syncthreads();
  for (int m = threadIdx.x; m < N_; m += 256) {
    const bf16* kr = qkv + qkvB + (size_t)m * 2304 + 768 + h * HD_;
    float d = 0.f;
#pragma unroll 8
    for (int kk = 0; kk < 64; kk++) d += q0[kk] * (float)kr[kk];
    sl[m] = d * 0.125f;
  }
  __syncthreads();
  float lm = -3e38f;
  for (int m = threadIdx.x; m < N_; m += 256) lm = fmaxf(lm, sl[m]);
  red[threadIdx.x] = lm; __syncthreads();
  for (int off = 128; off > 0; off >>= 1) {
    if (threadIdx.x < off)
      red[threadIdx.x] = fmaxf(red[threadIdx.x], red[threadIdx.x + off]);
    __syncthreads();
  }
  float mx = red[0];
  __syncthreads();
  float lsum = 0.f;
  for (int m = threadIdx.x; m < N_; m += 256) lsum += expf(sl[m] - mx);
  red[threadIdx.x] = lsum; __syncthreads();
  for (int off = 128; off > 0; off >>= 1) {
    if (threadIdx.x < off) red[threadIdx.x] += red[threadIdx.x + off];
    __syncthreads();
  }
  float denom = red[0];
  for (int m = threadIdx.x; m < N_; m += 256)
    atomicAdd(&cls[b * N_ + m], expf(sl[m] - mx) / denom * (1.f / H_));
}

// ---------------------------------------------------------------------------
// Gumbel perturb + top-288 (rank counting) + ascending compaction
// Forward-pass w = 1 + soft - stop_grad(soft) == 1.0 exactly.
// ---------------------------------------------------------------------------
__global__ __launch_bounds__(256)
void select_k(const float* __restrict__ cls, const float* __restrict__ gu,
              int* __restrict__ sel) {
  int b = blockIdx.x;
  __shared__ float pert[576];
  __shared__ int flag[576];
  for (int i = threadIdx.x; i < 576; i += 256) {
    float ca = cls[b * N_ + 1 + i];
    float lg = logf(fmaxf(ca, 1e-10f));
    float u = gu[b * 576 + i];
    float g = -logf(-logf(u + 1e-10f) + 1e-10f);
    pert[i] = (lg + g) * 0.5f;   // TAU = 2
  }
  __syncthreads();
  for (int i = threadIdx.x; i < 576; i += 256) {
    float pi = pert[i];
    int rank = 0;
    for (int j = 0; j < 576; j++) {
      float pj = pert[j];
      rank += (pj > pi) || (pj == pi && j < i);
    }
    flag[i] = (rank < KTOP) ? 1 : 0;
  }
  __syncthreads();
  if (threadIdx.x == 0) {
    int* sb = sel + b * NSEL;
    sb[0] = 0;
    int c = 1;
    for (int i = 0; i < 576; i++)
      if (flag[i]) sb[c++] = i + 1;
  }
}

// ---------------------------------------------------------------------------
// Gather selected tokens * sqrt(N / (1 + n_alpha))
// ---------------------------------------------------------------------------
__global__ __launch_bounds__(256)
void gather_k(const float* __restrict__ x2, const int* __restrict__ sel,
              float* __restrict__ out) {
  int b = blockIdx.x / NSEL;
  int s = blockIdx.x % NSEL;
  int tok = sel[b * NSEL + s];
  float sf = sqrtf((float)N_ / (float)NSEL);
  const float* src = x2 + ((size_t)b * N_ + tok) * D_;
  float* dst = out + ((size_t)b * NSEL + s) * D_;
  for (int c = threadIdx.x; c < D_; c += 256) dst[c] = src[c] * sf;
}

// ---------------------------------------------------------------------------
extern "C" void kernel_launch(void* const* d_in, const int* in_sizes, int n_in,
                              void* d_out, int out_size, void* d_ws, size_t ws_size,
                              hipStream_t stream) {
  const float* x     = (const float*)d_in[0];
  const float* gu    = (const float*)d_in[1];
  const float* ln1w  = (const float*)d_in[2];
  const float* ln1b  = (const float*)d_in[3];
  const float* qkvw  = (const float*)d_in[4];
  const float* qkvb  = (const float*)d_in[5];
  const float* projw = (const float*)d_in[6];
  const float* projb = (const float*)d_in[7];
  const float* ls1   = (const float*)d_in[8];
  const float* ln2w  = (const float*)d_in[9];
  const float* ln2b  = (const float*)d_in[10];
  const float* fc1w  = (const float*)d_in[11];
  const float* fc1b  = (const float*)d_in[12];
  const float* fc2w  = (const float*)d_in[13];
  const float* fc2b  = (const float*)d_in[14];
  const float* ls2   = (const float*)d_in[15];
  // d_in[16] = n_alpha (== 288, compile-time constant here)
  float* out = (float*)d_out;

  char* p = (char*)d_ws;
  auto alloc = [&](size_t bytes) -> char* {
    char* r = p;
    p += (bytes + 255) & ~(size_t)255;
    return r;
  };
  bf16*  wqkv  = (bf16*)alloc((size_t)768 * 2304 * 2);
  bf16*  wproj = (bf16*)alloc((size_t)768 * 768 * 2);
  bf16*  wfc1  = (bf16*)alloc((size_t)768 * 3072 * 2);
  bf16*  wfc2  = (bf16*)alloc((size_t)3072 * 768 * 2);
  bf16*  regA  = (bf16*)alloc((size_t)NTOK * 768 * 2);    // h -> ctx -> h2
  bf16*  regB  = (bf16*)alloc((size_t)NTOK * 3072 * 2);   // qkv -> a1
  bf16*  vT    = (bf16*)alloc((size_t)B_ * H_ * HD_ * NPAD * 2);
  float* x1    = (float*)alloc((size_t)NTOK * 768 * 4);
  float* x2    = (float*)alloc((size_t)NTOK * 768 * 4);
  float* cls   = (float*)alloc((size_t)B_ * N_ * 4);
  int*   sel   = (int*)alloc((size_t)B_ * NSEL * 4);

  (void)hipMemsetAsync(cls, 0, (size_t)B_ * N_ * 4, stream);
  cvt_k<<<1024, 256, 0, stream>>>(qkvw,  wqkv,  768 * 2304);
  cvt_k<<<512,  256, 0, stream>>>(projw, wproj, 768 * 768);
  cvt_k<<<1024, 256, 0, stream>>>(fc1w,  wfc1,  768 * 3072);
  cvt_k<<<1024, 256, 0, stream>>>(fc2w,  wfc2,  3072 * 768);

  // LN1 -> h (bf16)
  ln_k<<<NTOK, 256, 0, stream>>>(x, ln1w, ln1b, regA);
  // QKV GEMM: [NTOK,768] @ [768,2304] -> bf16
  gemm_bf16_k<0><<<dim3(2304 / 64, (NTOK + 127) / 128), 256, 0, stream>>>(
      regA, wqkv, qkvb, nullptr, nullptr, regB, nullptr, NTOK, 768, 2304);
  // V transpose (padded) for guard-free attention B-fragments
  vtr_k<<<B_ * H_, 256, 0, stream>>>(regB, vT);
  // Flash attention -> ctx (bf16, reuses regA)
  attn_k<<<B_ * H_ * 5, 256, 0, stream>>>(regB, vT, regA);
  // CLS attention distribution (mean over heads)
  clsattn_k<<<B_ * H_, 256, 0, stream>>>(regB, cls);
  // proj GEMM + residual: x1 = x + ls1 * (ctx@Wp + bp)
  gemm_bf16_k<1><<<dim3(768 / 64, (NTOK + 127) / 128), 256, 0, stream>>>(
      regA, wproj, projb, x, ls1, nullptr, x1, NTOK, 768, 768);
  // LN2 -> h2 (bf16, reuses regA)
  ln_k<<<NTOK, 256, 0, stream>>>(x1, ln2w, ln2b, regA);
  // fc1 GEMM + exact GELU -> a1 (bf16, reuses regB)
  gemm_bf16_k<2><<<dim3(3072 / 64, (NTOK + 127) / 128), 256, 0, stream>>>(
      regA, wfc1, fc1b, nullptr, nullptr, regB, nullptr, NTOK, 768, 3072);
  // fc2 GEMM + residual: x2 = x1 + ls2 * (a1@W2 + b2)
  gemm_bf16_k<1><<<dim3(768 / 64, (NTOK + 127) / 128), 256, 0, stream>>>(
      regB, wfc2, fc2b, x1, ls2, nullptr, x2, NTOK, 3072, 768);
  // Gumbel top-k selection + gather
  select_k<<<B_, 256, 0, stream>>>(cls, gu, sel);
  gather_k<<<B_ * NSEL, 256, 0, stream>>>(x2, sel, out);
}
